// POSTagger_74448963108992
// MI455X (gfx1250) — matrix operand order repaired and checked by reading, not compiled, not run on hardware
//
#include <hip/hip_runtime.h>
#include <math.h>

// ---------------- model constants ----------------
#define BATCH   2
#define T_LEN   4096
#define DIMM    512
#define NHEAD   8
#define DHEAD   64
#define NHASH   8
#define NBKT    64          // buckets per hash round
#define NBIN    512         // NHASH * NBKT
#define NCHK    512         // (NHASH*T)/64 sorted chunks per bh
#define BHTOT   16          // BATCH * NHEAD
#define MROWS   8192        // BATCH * T
#define NTOT    32768       // NHASH * T
#define TAGS    17
#define FFD     2048
#define MASKV   (-5.0e4f)

typedef __attribute__((ext_vector_type(16))) __bf16 v16bf;
typedef __attribute__((ext_vector_type(8)))  __bf16 v8bf;
typedef __attribute__((ext_vector_type(8)))  float  v8f;

// ---------------- WMMA fragment loaders (CDNA5 16x16x32 bf16 layouts) ----
// A: 16x32 row-major; lane l holds row M=l%16. Per lane the fragment is two
// contiguous 8-element runs: K = 8g+0..7 and K = 16+8g+0..7  -> b128 loads.
__device__ __forceinline__ v16bf load_a_frag_bf(const __bf16* src, int ld, int row0, int k0, int lane) {
  int m = lane & 15, g = lane >> 4;
  const __bf16* pr = src + (size_t)(row0 + m) * ld + k0 + 8 * g;
  v16bf a;
  *(v8bf*)&a       = *(const v8bf*)pr;
  *((v8bf*)&a + 1) = *(const v8bf*)(pr + 16);
  return a;
}
// Same fragment from f32 source (probs in LDS) with convert.
__device__ __forceinline__ v16bf load_a_frag_f32(const float* src, int ld, int row0, int k0, int lane) {
  int m = lane & 15, g = lane >> 4;
  const float* pr = src + (size_t)(row0 + m) * ld + k0 + 8 * g;
  v16bf a;
#pragma unroll
  for (int e = 0; e < 8; ++e) {
    a[e]     = (__bf16)pr[e];
    a[8 + e] = (__bf16)pr[16 + e];
  }
  return a;
}
// B: 32x16, sourced from a TRANSPOSED buffer Xt[N][K] (B[k][n] = Xt[n][k]).
// Lane l holds col N=l%16; two 8-element runs K = 8g+0..7 and 16+8g+0..7 -> b128 loads.
__device__ __forceinline__ v16bf load_bT_frag_bf(const __bf16* src, int ld, int k0, int n0, int lane) {
  int n = lane & 15, g = lane >> 4;
  const __bf16* pr = src + (size_t)(n0 + n) * ld + k0 + 8 * g;
  v16bf b;
  *(v8bf*)&b       = *(const v8bf*)pr;
  *((v8bf*)&b + 1) = *(const v8bf*)(pr + 16);
  return b;
}

__device__ __forceinline__ float gelu_tanh(float x) {
  return 0.5f * x * (1.0f + tanhf(0.7978845608028654f * (x + 0.044715f * x * x * x)));
}

// ---------------- GEMM: C[M,N] = epilogue(A[M,K] @ Wt[N,K]^T) -------------
// Each wave: 16 rows x 32 cols (2 WMMA tiles sharing the A fragment).
// Software-pipelined: k+32 fragments are in flight while k's WMMAs execute.
// modes: 0 plain; 1 +bias; 2 gelu(+bias); 3 +res; 4 +res+bias
// Cbf (optional) = fused bf16 output; C (optional) = f32 output.
__global__ __launch_bounds__(256) void gemm_bf16(
    const __bf16* __restrict__ A, const __bf16* __restrict__ Wt,
    float* __restrict__ C, __bf16* __restrict__ Cbf,
    const float* __restrict__ bias, const float* __restrict__ res,
    int M, int N, int K, int mode)
{
  int lane = threadIdx.x & 31;
  int w    = threadIdx.x >> 5;
  int row0    = blockIdx.y * 64 + (w & 3) * 16;
  int colbase = blockIdx.x * 64 + (w >> 2) * 32;
  v8f acc0 = {}, acc1 = {};
  v16bf a  = load_a_frag_bf(A, K, row0, 0, lane);
  v16bf b0 = load_bT_frag_bf(Wt, K, 0, colbase, lane);
  v16bf b1 = load_bT_frag_bf(Wt, K, 0, colbase + 16, lane);
  for (int k = 0; k < K; k += 32) {
    v16bf an = a, bn0 = b0, bn1 = b1;
    int kn = k + 32;
    if (kn < K) {
      __builtin_prefetch(A + (size_t)row0 * K + kn + 224, 0, 3);
      an  = load_a_frag_bf(A, K, row0, kn, lane);
      bn0 = load_bT_frag_bf(Wt, K, kn, colbase, lane);
      bn1 = load_bT_frag_bf(Wt, K, kn, colbase + 16, lane);
    }
    acc0 = __builtin_amdgcn_wmma_f32_16x16x32_bf16(false, a, false, b0, (short)0, acc0, false, false);
    acc1 = __builtin_amdgcn_wmma_f32_16x16x32_bf16(false, a, false, b1, (short)0, acc1, false, false);
    a = an; b0 = bn0; b1 = bn1;
  }
  int g = lane >> 4, n = lane & 15;
#pragma unroll
  for (int tt = 0; tt < 2; ++tt) {
    int col = colbase + 16 * tt + n;
#pragma unroll
    for (int r = 0; r < 8; ++r) {
      int row = row0 + r + 8 * g;
      float v = (tt == 0) ? acc0[r] : acc1[r];
      if (mode == 1)      v += bias[col];
      else if (mode == 2) v = gelu_tanh(v + bias[col]);
      else if (mode == 3) v += res[(size_t)row * N + col];
      else if (mode == 4) v += res[(size_t)row * N + col] + bias[col];
      if (Cbf) Cbf[(size_t)row * N + col] = (__bf16)v;
      if (C)   C[(size_t)row * N + col] = v;
    }
  }
}

// ---------------- embedding + axial positions ----------------
__global__ void embed_kernel(const int* __restrict__ X, const float* __restrict__ emb,
                             const float* __restrict__ ax1, const float* __restrict__ ax2,
                             float* __restrict__ x1, float* __restrict__ x2)
{
  int tid = blockIdx.x * blockDim.x + threadIdx.x;
  if (tid >= MROWS * DIMM) return;
  int m = tid >> 9, col = tid & 511;
  int t = m & (T_LEN - 1);
  int tok = X[m];
  float v = emb[(size_t)tok * DIMM + col];
  v += (col < 256) ? ax1[(t >> 6) * 256 + col] : ax2[(t & 63) * 256 + (col - 256)];
  x1[tid] = v; x2[tid] = v;
}

// ---------------- LayerNorm (optionally averaging two streams) ----------
__global__ __launch_bounds__(256) void ln_kernel(
    const float* __restrict__ x, const float* __restrict__ xavg,
    const float* __restrict__ gma, const float* __restrict__ bta,
    __bf16* __restrict__ out, int M)
{
  int lane = threadIdx.x & 31, w = threadIdx.x >> 5;
  int row = blockIdx.x * 8 + w;
  if (row >= M) return;
  const float* p  = x + (size_t)row * DIMM;
  const float* p2 = xavg ? xavg + (size_t)row * DIMM : nullptr;
  float v[16];
  float s = 0.f;
#pragma unroll
  for (int k = 0; k < 16; ++k) {
    float t = p[lane + 32 * k];
    if (p2) t = 0.5f * (t + p2[lane + 32 * k]);
    v[k] = t; s += t;
  }
#pragma unroll
  for (int o = 16; o; o >>= 1) s += __shfl_xor(s, o, 32);
  float mean = s * (1.0f / DIMM);
  float vs = 0.f;
#pragma unroll
  for (int k = 0; k < 16; ++k) { float d = v[k] - mean; vs += d * d; }
#pragma unroll
  for (int o = 16; o; o >>= 1) vs += __shfl_xor(vs, o, 32);
  float inv = rsqrtf(vs * (1.0f / DIMM) + 1e-5f);
#pragma unroll
  for (int k = 0; k < 16; ++k) {
    int col = lane + 32 * k;
    out[(size_t)row * DIMM + col] = (__bf16)((v[k] - mean) * inv * gma[col] + bta[col]);
  }
}

// weight convert WITH transpose: in[K][N] (f32, row-major) -> out[N][K] (bf16)
__global__ void f2bfT_kernel(const float* __restrict__ in, __bf16* __restrict__ out, int K, int N) {
  int tid = blockIdx.x * blockDim.x + threadIdx.x;
  if (tid >= K * N) return;
  int k = tid / N, n = tid - k * N;
  out[(size_t)n * K + k] = (__bf16)in[tid];
}
__global__ void zero_i32_kernel(int* __restrict__ p, int n) {
  for (int i = blockIdx.x * blockDim.x + threadIdx.x; i < n; i += gridDim.x * blockDim.x)
    p[i] = 0;
}

// ---------------- LSH bucketing: argmax over [R, -R] rotations -----------
__global__ void bucket_kernel(const float* __restrict__ qk, const float* __restrict__ rot,
                              int* __restrict__ buckets)
{
  int tid = blockIdx.x * blockDim.x + threadIdx.x;
  if (tid >= BHTOT * NHASH * T_LEN) return;
  int t  = tid & (T_LEN - 1);
  int hs = (tid >> 12) & (NHASH - 1);
  int bh = tid >> 15;
  int b = bh >> 3, hd = bh & 7;
  const float4* q4 = (const float4*)(qk + (size_t)(b * T_LEN + t) * DIMM + hd * DHEAD);
  float qv[64];
#pragma unroll
  for (int j = 0; j < 16; ++j) {
    float4 f = q4[j];
    qv[4*j] = f.x; qv[4*j+1] = f.y; qv[4*j+2] = f.z; qv[4*j+3] = f.w;
  }
  float sr[32];
  for (int r = 0; r < 32; ++r) {
    float acc = 0.f;
    for (int d = 0; d < 64; ++d) acc += qv[d] * rot[(d * NHASH + hs) * 32 + r];
    sr[r] = acc;
  }
  float best = sr[0]; int bi = 0;
  for (int r = 1; r < 32; ++r) if (sr[r] > best) { best = sr[r]; bi = r; }
  for (int r = 0; r < 32; ++r) if (-sr[r] > best) { best = -sr[r]; bi = 32 + r; }
  buckets[(size_t)bh * NTOT + hs * T_LEN + t] = hs * NBKT + bi;
}

// ---------------- stable counting sort (== argsort of t*bucket + pos) ----
__global__ void count_kernel(const int* __restrict__ buckets, int* __restrict__ counts)
{
  int tid = blockIdx.x * blockDim.x + threadIdx.x;
  if (tid >= BHTOT * NCHK) return;
  int bh = tid >> 9, ch = tid & 511;
  for (int j = 0; j < 64; ++j) {
    int f  = ch * 64 + j;
    int bv = buckets[(size_t)bh * NTOT + f];
    int idx = ((bh * NBIN + bv) * NCHK) + ch;
    counts[idx] = counts[idx] + 1;                 // unique writer per (bh,*,ch)
  }
}

__device__ __forceinline__ int block_incl_scan512(int v, int* sh) {
  int tid = threadIdx.x;
  sh[tid] = v; __syncthreads();
  for (int off = 1; off < 512; off <<= 1) {
    int t = (tid >= off) ? sh[tid - off] : 0;
    __syncthreads();
    sh[tid] += t;
    __syncthreads();
  }
  return sh[tid];
}

__global__ __launch_bounds__(512) void bucket_scan_kernel(const int* __restrict__ counts,
                                                          int* __restrict__ bstart)
{
  __shared__ int sh[512];
  int bh = blockIdx.x, bv = threadIdx.x;
  int s = 0;
  for (int c = 0; c < NCHK; ++c) s += counts[((bh * NBIN + bv) * NCHK) + c];
  int incl = block_incl_scan512(s, sh);
  bstart[bh * NBIN + bv] = incl - s;               // exclusive
}

__global__ __launch_bounds__(512) void chunk_scan_kernel(const int* __restrict__ counts,
                                                         const int* __restrict__ bstart,
                                                         int* __restrict__ offs)
{
  __shared__ int sh[512];
  int bh = blockIdx.x >> 9, bv = blockIdx.x & 511, ch = threadIdx.x;
  int v = counts[((bh * NBIN + bv) * NCHK) + ch];
  int incl = block_incl_scan512(v, sh);
  offs[((bh * NBIN + bv) * NCHK) + ch] = bstart[bh * NBIN + bv] + incl - v;
}

__global__ void scatter_kernel(const int* __restrict__ buckets, int* __restrict__ offs,
                               int* __restrict__ sticker)
{
  int tid = blockIdx.x * blockDim.x + threadIdx.x;
  if (tid >= BHTOT * NCHK) return;
  int bh = tid >> 9, ch = tid & 511;
  for (int j = 0; j < 64; ++j) {
    int f   = ch * 64 + j;
    int bv  = buckets[(size_t)bh * NTOT + f];
    int idx = ((bh * NBIN + bv) * NCHK) + ch;
    int pos = offs[idx];
    offs[idx] = pos + 1;
    sticker[(size_t)bh * NTOT + pos] = f;
  }
}

// ---------------- bucketed attention (WMMA) ----------------
// grid (NCHK, BHTOT), block 128 (4 waves). Dynamic LDS ~73KB.
__global__ __launch_bounds__(128) void lsh_attn_kernel(
    const float* __restrict__ qk, const float* __restrict__ vbuf,
    const int* __restrict__ sticker,
    float* __restrict__ o_rounds, float* __restrict__ lrounds)
{
  extern __shared__ char smem_raw[];
  __bf16* qs   = (__bf16*)smem_raw;          // [64][64]
  __bf16* ks   = qs + 64 * 64;               // [128][64] normalized keys (row-major)
  __bf16* vsT  = ks + 128 * 64;              // [64][128] values TRANSPOSED (dim-major)
  float*  dots = (float*)(vsT + 64 * 128);   // [64][128]
  int*    qpos = (int*)(dots + 64 * 128);    // 64
  int*    kpos = qpos + 64;                  // 128
  int*    fidx = kpos + 128;                 // 64

  int c  = blockIdx.x;
  int bh = blockIdx.y;
  int b = bh >> 3, hd = bh & 7;
  int tid = threadIdx.x;
  int lane = tid & 31, w = tid >> 5;
  int cprev = (c + NCHK - 1) & (NCHK - 1);

  if (tid < 64) {
    int f = sticker[(size_t)bh * NTOT + c * 64 + tid];
    fidx[tid] = f;
    int tp = f & (T_LEN - 1);
    qpos[tid] = tp;
    kpos[tid] = tp;
  } else {
    int i = tid - 64;
    int f = sticker[(size_t)bh * NTOT + cprev * 64 + i];
    kpos[64 + i] = f & (T_LEN - 1);
  }
  __syncthreads();

  { // gather q -> bf16 (float4 loads; 2 threads per row)
    int i = tid >> 1, half = tid & 1;
    const float4* src = (const float4*)(qk + (size_t)(b * T_LEN + qpos[i]) * DIMM
                                           + hd * DHEAD + half * 32);
    __bf16* dst = qs + i * 64 + half * 32;
#pragma unroll
    for (int j = 0; j < 8; ++j) {
      float4 f = src[j];
      dst[4*j]   = (__bf16)f.x; dst[4*j+1] = (__bf16)f.y;
      dst[4*j+2] = (__bf16)f.z; dst[4*j+3] = (__bf16)f.w;
    }
  }
  { // gather normalized k (row-major) and v (transposed), one row per thread
    int i = tid;
    int grow = b * T_LEN + kpos[i];
    const float4* sq4 = (const float4*)(qk   + (size_t)grow * DIMM + hd * DHEAD);
    const float4* sv4 = (const float4*)(vbuf + (size_t)grow * DIMM + hd * DHEAD);
    float tmp[64]; float ss = 0.f;
#pragma unroll
    for (int j = 0; j < 16; ++j) {
      float4 f = sq4[j];
      tmp[4*j] = f.x; tmp[4*j+1] = f.y; tmp[4*j+2] = f.z; tmp[4*j+3] = f.w;
      ss += f.x*f.x + f.y*f.y + f.z*f.z + f.w*f.w;
    }
    float inv = rsqrtf(ss);
#pragma unroll
    for (int j = 0; j < 64; ++j) ks[i * 64 + j] = (__bf16)(tmp[j] * inv);
#pragma unroll
    for (int j = 0; j < 16; ++j) {
      float4 f = sv4[j];
      vsT[(4*j)   * 128 + i] = (__bf16)f.x;
      vsT[(4*j+1) * 128 + i] = (__bf16)f.y;
      vsT[(4*j+2) * 128 + i] = (__bf16)f.z;
      vsT[(4*j+3) * 128 + i] = (__bf16)f.w;
    }
  }
  __syncthreads();

  // dots = (q @ k^T) / sqrt(DH), masked where q_pos == k_pos
  for (int nt = 0; nt < 8; ++nt) {
    v8f acc = {};
    for (int kk = 0; kk < 64; kk += 32) {
      v16bf a  = load_a_frag_bf(qs, 64, w * 16, kk, lane);
      v16bf bf = load_bT_frag_bf(ks, 64, kk, nt * 16, lane);   // B = k^T
      acc = __builtin_amdgcn_wmma_f32_16x16x32_bf16(false, a, false, bf, (short)0, acc, false, false);
    }
    int g = lane >> 4, n = lane & 15;
    int col = nt * 16 + n;
#pragma unroll
    for (int r = 0; r < 8; ++r) {
      int row = w * 16 + r + 8 * g;
      float d = acc[r] * 0.125f;                  // 1/sqrt(64)
      if (qpos[row] == kpos[col]) d = MASKV;
      dots[row * 128 + col] = d;
    }
  }
  __syncthreads();

  // softmax per row; store logsumexp per (round, t)
  if (tid < 64) {
    float mx = -3.4e38f;
    for (int j = 0; j < 128; ++j) mx = fmaxf(mx, dots[tid * 128 + j]);
    float s = 0.f;
    for (int j = 0; j < 128; ++j) s += __expf(dots[tid * 128 + j] - mx);
    float lse = mx + __logf(s);
    for (int j = 0; j < 128; ++j) dots[tid * 128 + j] = __expf(dots[tid * 128 + j] - lse);
    int f = fidx[tid];
    int rnd = f >> 12, tp = f & (T_LEN - 1);
    lrounds[(size_t)(bh * NHASH + rnd) * T_LEN + tp] = lse;
  }
  __syncthreads();

  // out = probs @ v, scattered to per-round unsorted positions
  for (int nt = 0; nt < 4; ++nt) {
    v8f acc = {};
    for (int kk = 0; kk < 128; kk += 32) {
      v16bf a  = load_a_frag_f32(dots, 128, w * 16, kk, lane);
      v16bf bf = load_bT_frag_bf(vsT, 128, kk, nt * 16, lane); // vsT[n][k] == v[k][n]
      acc = __builtin_amdgcn_wmma_f32_16x16x32_bf16(false, a, false, bf, (short)0, acc, false, false);
    }
    int g = lane >> 4, n = lane & 15;
#pragma unroll
    for (int r = 0; r < 8; ++r) {
      int row = w * 16 + r + 8 * g;
      int f = fidx[row];
      int rnd = f >> 12, tp = f & (T_LEN - 1);
      o_rounds[((size_t)(bh * NHASH + rnd) * T_LEN + tp) * DHEAD + nt * 16 + n] = acc[r];
    }
  }
}

// ------- combine NHASH rounds, merge heads, emit bf16 for the Wo GEMM -----
__global__ void combine_kernel(const float* __restrict__ o_rounds,
                               const float* __restrict__ lrounds,
                               __bf16* __restrict__ attbf)
{
  int tid = blockIdx.x * blockDim.x + threadIdx.x;
  if (tid >= BHTOT * T_LEN) return;
  int bh = tid >> 12, t = tid & (T_LEN - 1);
  float lg[NHASH]; float mx = -3.4e38f;
#pragma unroll
  for (int r = 0; r < NHASH; ++r) {
    lg[r] = lrounds[(size_t)(bh * NHASH + r) * T_LEN + t];
    mx = fmaxf(mx, lg[r]);
  }
  float wts[NHASH]; float s = 0.f;
#pragma unroll
  for (int r = 0; r < NHASH; ++r) { wts[r] = __expf(lg[r] - mx); s += wts[r]; }
  float inv = 1.0f / s;
  int b = bh >> 3, hd = bh & 7;
  __bf16* dst = attbf + (size_t)(b * T_LEN + t) * DIMM + hd * DHEAD;
  for (int d = 0; d < DHEAD; ++d) {
    float acc = 0.f;
#pragma unroll
    for (int r = 0; r < NHASH; ++r)
      acc += o_rounds[((size_t)(bh * NHASH + r) * T_LEN + t) * DHEAD + d] * wts[r];
    dst[d] = (__bf16)(acc * inv);
  }
}

// ---------------- final 512 -> 17 projection ----------------
__global__ void final_proj_kernel(const __bf16* __restrict__ A, const float* __restrict__ Wout,
                                  const float* __restrict__ bout, float* __restrict__ out)
{
  int tid = blockIdx.x * blockDim.x + threadIdx.x;
  if (tid >= MROWS * TAGS) return;
  int m = tid / TAGS, tag = tid - m * TAGS;
  const __bf16* a = A + (size_t)m * DIMM;
  float acc = bout[tag];
  for (int k = 0; k < DIMM; ++k) acc += (float)a[k] * Wout[k * TAGS + tag];
  out[tid] = acc;
}

// ==================== host orchestration ====================
extern "C" void kernel_launch(void* const* d_in, const int* in_sizes, int n_in,
                              void* d_out, int out_size, void* d_ws, size_t ws_size,
                              hipStream_t stream)
{
  const int*   X    = (const int*)  d_in[0];
  const float* temb = (const float*)d_in[1];
  const float* ax1  = (const float*)d_in[2];
  const float* ax2  = (const float*)d_in[3];
  const float* Wqk  = (const float*)d_in[4];
  const float* Wv   = (const float*)d_in[5];
  const float* Wo   = (const float*)d_in[6];
  const float* ln1g = (const float*)d_in[7];
  const float* ln1b = (const float*)d_in[8];
  const float* W1   = (const float*)d_in[9];
  const float* b1   = (const float*)d_in[10];
  const float* W2   = (const float*)d_in[11];
  const float* b2   = (const float*)d_in[12];
  const float* ln2g = (const float*)d_in[13];
  const float* ln2b = (const float*)d_in[14];
  const float* lnfg = (const float*)d_in[15];
  const float* lnfb = (const float*)d_in[16];
  const float* Wout = (const float*)d_in[17];
  const float* bout = (const float*)d_in[18];
  const float* rot  = (const float*)d_in[19];
  float* out = (float*)d_out;

  char* base = (char*)d_ws; size_t off = 0;
  auto alloc = [&](size_t bytes) -> void* {
    void* r = base + off; off += (bytes + 255) & ~(size_t)255; return r;
  };
  float*  x1      = (float*) alloc((size_t)MROWS * DIMM * 4);
  float*  x2      = (float*) alloc((size_t)MROWS * DIMM * 4);
  __bf16* lnbf    = (__bf16*)alloc((size_t)MROWS * DIMM * 2);
  __bf16* attbf   = (__bf16*)alloc((size_t)MROWS * DIMM * 2);
  float*  qkbuf   = (float*) alloc((size_t)MROWS * DIMM * 4);
  float*  vbuf    = (float*) alloc((size_t)MROWS * DIMM * 4);
  __bf16* hbf     = (__bf16*)alloc((size_t)MROWS * FFD * 2);
  __bf16* wbf     = (__bf16*)alloc((size_t)DIMM * FFD * 2);   // transposed weights
  int*    buckets = (int*)   alloc((size_t)BHTOT * NTOT * 4);
  int*    counts  = (int*)   alloc((size_t)BHTOT * NBIN * NCHK * 4);
  int*    offs    = (int*)   alloc((size_t)BHTOT * NBIN * NCHK * 4);
  int*    bstart  = (int*)   alloc((size_t)BHTOT * NBIN * 4);
  int*    sticker = (int*)   alloc((size_t)BHTOT * NTOT * 4);
  float*  orounds = (float*) alloc((size_t)BHTOT * NHASH * T_LEN * DHEAD * 4);
  float*  lrounds = (float*) alloc((size_t)BHTOT * NHASH * T_LEN * 4);

  auto cvtT = [&](const float* in, __bf16* o, int K, int N) {   // -> [N][K]
    f2bfT_kernel<<<(K * N + 255) / 256, 256, 0, stream>>>(in, o, K, N);
  };
  auto gemm = [&](const __bf16* A, const __bf16* Wt, float* C, __bf16* Cbf,
                  int M, int N, int K, int mode, const float* bias, const float* res) {
    gemm_bf16<<<dim3(N / 64, M / 64), 256, 0, stream>>>(A, Wt, C, Cbf, bias, res, M, N, K, mode);
  };

  // embedding + axial positions
  embed_kernel<<<(MROWS * DIMM + 255) / 256, 256, 0, stream>>>(X, temb, ax1, ax2, x1, x2);

  const int SMEM_ATTN = (64 * 64 + 128 * 64 + 64 * 128) * 2 + 64 * 128 * 4 + (64 + 128 + 64) * 4;

  for (int l = 0; l < 2; ++l) {
    // ---- attention half: x1 += Attn(LN(x2)) @ Wo ----
    ln_kernel<<<MROWS / 8, 256, 0, stream>>>(x2, nullptr, ln1g + l * DIMM, ln1b + l * DIMM, lnbf, MROWS);
    cvtT(Wqk + (size_t)l * DIMM * DIMM, wbf, DIMM, DIMM);
    gemm(lnbf, wbf, qkbuf, nullptr, MROWS, DIMM, DIMM, 0, nullptr, nullptr);
    cvtT(Wv + (size_t)l * DIMM * DIMM, wbf, DIMM, DIMM);
    gemm(lnbf, wbf, vbuf, nullptr, MROWS, DIMM, DIMM, 0, nullptr, nullptr);

    // LSH bucketing + stable counting sort
    bucket_kernel<<<(BHTOT * NHASH * T_LEN + 255) / 256, 256, 0, stream>>>(
        qkbuf, rot + (size_t)l * DHEAD * NHASH * 32, buckets);
    zero_i32_kernel<<<4096, 256, 0, stream>>>(counts, BHTOT * NBIN * NCHK);
    count_kernel<<<(BHTOT * NCHK + 255) / 256, 256, 0, stream>>>(buckets, counts);
    bucket_scan_kernel<<<BHTOT, 512, 0, stream>>>(counts, bstart);
    chunk_scan_kernel<<<BHTOT * NBIN, 512, 0, stream>>>(counts, bstart, offs);
    scatter_kernel<<<(BHTOT * NCHK + 255) / 256, 256, 0, stream>>>(buckets, offs, sticker);

    // bucketed attention + round combine (combine emits bf16 directly)
    lsh_attn_kernel<<<dim3(NCHK, BHTOT), 128, SMEM_ATTN, stream>>>(
        qkbuf, vbuf, sticker, orounds, lrounds);
    combine_kernel<<<(BHTOT * T_LEN + 255) / 256, 256, 0, stream>>>(orounds, lrounds, attbf);

    cvtT(Wo + (size_t)l * DIMM * DIMM, wbf, DIMM, DIMM);
    gemm(attbf, wbf, x1, nullptr, MROWS, DIMM, DIMM, 3, nullptr, x1);   // x1 += attn_out @ Wo

    // ---- FF half: x2 += gelu(LN(x1) @ W1 + b1) @ W2 + b2 ----
    ln_kernel<<<MROWS / 8, 256, 0, stream>>>(x1, nullptr, ln2g + l * DIMM, ln2b + l * DIMM, lnbf, MROWS);
    cvtT(W1 + (size_t)l * DIMM * FFD, wbf, DIMM, FFD);
    gemm(lnbf, wbf, nullptr, hbf, MROWS, FFD, DIMM, 2, b1 + l * FFD, nullptr); // bf16 hidden, fused
    cvtT(W2 + (size_t)l * FFD * DIMM, wbf, FFD, DIMM);
    gemm(hbf, wbf, x2, nullptr, MROWS, DIMM, FFD, 4, b2 + l * DIMM, x2);       // x2 += h@W2+b2
  }

  // final: LN(0.5*(x1+x2)) @ Wout + bout
  ln_kernel<<<MROWS / 8, 256, 0, stream>>>(x1, x2, lnfg, lnfb, lnbf, MROWS);
  final_proj_kernel<<<(MROWS * TAGS + 255) / 256, 256, 0, stream>>>(lnbf, Wout, bout, out);
}
